// Network_84593675862038
// MI455X (gfx1250) — matrix operand order repaired and checked by reading, not compiled, not run on hardware
//
#include <hip/hip_runtime.h>
#include <hip/hip_bf16.h>

typedef __attribute__((ext_vector_type(16))) _Float16 v16h;
typedef __attribute__((ext_vector_type(8)))  float    v8f;

union HFrag { v16h v; uint4 q[2]; };

#define BN_EPS 1e-5f

// ---------------------------------------------------------------------------
// feats0: (B,3,N) channel-first copy of pcs coordinates
// ---------------------------------------------------------------------------
__global__ void feats0_kernel(const float* __restrict__ pcs, float* __restrict__ f0, int N) {
  int idx = blockIdx.x * blockDim.x + threadIdx.x;
  int tot = 8 * N;
  if (idx >= tot) return;
  int b = idx / N, n = idx % N;
#pragma unroll
  for (int c = 0; c < 3; ++c)
    f0[((size_t)b * 3 + c) * N + n] = pcs[((size_t)b * N + n) * 3 + c];
}

// ---------------------------------------------------------------------------
// Furthest point sampling: one block per batch, dists kept in LDS.
// Emits new_xyz directly (gather of selected indices).
// ---------------------------------------------------------------------------
__global__ void fps_kernel(const float* __restrict__ xyz, int N, int npoint,
                           float* __restrict__ out) {
  extern __shared__ char smem_raw[];
  float* dists = (float*)smem_raw;
  float* rv = dists + N;
  int*   ri = (int*)(rv + 256);
  int b = blockIdx.x, tid = threadIdx.x;
  const float* base = xyz + (size_t)b * N * 3;
  for (int i = tid; i < N; i += 256) dists[i] = 1e10f;
  __syncthreads();
  int far = 0;
  for (int it = 0; it < npoint; ++it) {
    float cx = base[far * 3 + 0], cy = base[far * 3 + 1], cz = base[far * 3 + 2];
    if (tid == 0) {
      float* o = out + ((size_t)b * npoint + it) * 3;
      o[0] = cx; o[1] = cy; o[2] = cz;
    }
    float bm = -1.f; int bi = 0x7fffffff;
    for (int i = tid; i < N; i += 256) {
      float dx = base[i * 3 + 0] - cx;
      float dy = base[i * 3 + 1] - cy;
      float dz = base[i * 3 + 2] - cz;
      float d = dx * dx + dy * dy + dz * dz;
      float cur = dists[i];
      if (d < cur) { cur = d; dists[i] = cur; }
      if (cur > bm) { bm = cur; bi = i; }
    }
    rv[tid] = bm; ri[tid] = bi;
    __syncthreads();
    for (int o = 128; o > 0; o >>= 1) {
      if (tid < o) {
        if (rv[tid + o] > rv[tid] ||
            (rv[tid + o] == rv[tid] && ri[tid + o] < ri[tid])) {
          rv[tid] = rv[tid + o]; ri[tid] = ri[tid + o];
        }
      }
      __syncthreads();
    }
    far = ri[0];
    __syncthreads();
  }
}

// ---------------------------------------------------------------------------
// Ball query + grouping: one thread per (b,s). Writes f16 X rows (M, Kp):
// [rel_xyz(3), gathered feats(Cf), zero pad].
// Key trick: reference top_k over index-keys == first `ns` in-radius indices,
// padded with the first hit.
// ---------------------------------------------------------------------------
__global__ void ball_group_kernel(const float* __restrict__ xyz,
                                  const float* __restrict__ new_xyz,
                                  const float* __restrict__ feats,
                                  int Nprev, int S, int ns, float r2,
                                  int Cf, int Kp, _Float16* __restrict__ Xh) {
  int q = blockIdx.x * blockDim.x + threadIdx.x;
  if (q >= 8 * S) return;
  int b = q / S;
  const float* nc = new_xyz + (size_t)q * 3;
  float cx = nc[0], cy = nc[1], cz = nc[2];
  int idx[32];
  int cnt = 0, first = -1;
  const float* base = xyz + (size_t)b * Nprev * 3;
  for (int n = 0; n < Nprev && cnt < ns; ++n) {
    float dx = base[n * 3 + 0] - cx;
    float dy = base[n * 3 + 1] - cy;
    float dz = base[n * 3 + 2] - cz;
    float d2 = dx * dx + dy * dy + dz * dz;
    if (d2 < r2) { if (first < 0) first = n; idx[cnt++] = n; }
  }
  if (first < 0) first = 0;
  for (; cnt < ns; ++cnt) idx[cnt] = first;
  for (int j = 0; j < ns; ++j) {
    int n = idx[j];
    _Float16* row = Xh + ((size_t)q * ns + j) * Kp;
    row[0] = (_Float16)(base[n * 3 + 0] - cx);
    row[1] = (_Float16)(base[n * 3 + 1] - cy);
    row[2] = (_Float16)(base[n * 3 + 2] - cz);
    for (int c = 0; c < Cf; ++c)
      row[3 + c] = (_Float16)feats[((size_t)b * Cf + c) * Nprev + n];
    for (int k = 3 + Cf; k < Kp; ++k) row[k] = (_Float16)0.f;
  }
}

// ---------------------------------------------------------------------------
// Pack f32 weights (C, Cin) -> f16 K-padded (C, Kp)
// ---------------------------------------------------------------------------
__global__ void pack_w_kernel(const float* __restrict__ W, int C, int Cin, int Kp,
                              _Float16* __restrict__ Wh) {
  int idx = blockIdx.x * blockDim.x + threadIdx.x;
  if (idx >= C * Kp) return;
  int c = idx / Kp, k = idx % Kp;
  Wh[idx] = (k < Cin) ? (_Float16)W[(size_t)c * Cin + k] : (_Float16)0.f;
}

// ---------------------------------------------------------------------------
// WMMA GEMM: Y(C,M) = Wh(C,Kp) x Xh(M,Kp)^T  (K contiguous in both operands)
// One wave32 per 16(C) x 64(M) macro-tile; A fragment reused over 4 N-tiles.
// Fragment layouts per CDNA5 ISA 7.12.2 (f16 A 16x32, C/D 16x16 f32).
// ---------------------------------------------------------------------------
__global__ __launch_bounds__(32)
void wmma_gemm_kernel(const _Float16* __restrict__ Wh,
                      const _Float16* __restrict__ Xh,
                      float* __restrict__ Y, int C, int Kp, int M) {
  const int lane = threadIdx.x;
  const int m0 = blockIdx.x * 64;
  const int c0 = blockIdx.y * 16;
  const int hs = lane >> 4;        // lane half select
  const int ln = lane & 15;
  v8f zero = {0.f, 0.f, 0.f, 0.f, 0.f, 0.f, 0.f, 0.f};
  v8f acc0 = zero, acc1 = zero, acc2 = zero, acc3 = zero;
  const _Float16* wrow = Wh + (size_t)(c0 + ln) * Kp;
  for (int k0 = 0; k0 < Kp; k0 += 32) {
    HFrag a;
    a.q[0] = *(const uint4*)(wrow + k0 + hs * 8);        // K = k0+hs*8 .. +7
    a.q[1] = *(const uint4*)(wrow + k0 + 16 + hs * 8);   // K = k0+16+hs*8 .. +7
    HFrag bb;
    const _Float16* x0 = Xh + (size_t)(m0 + ln) * Kp + k0 + hs * 8;
#pragma unroll
    for (int t = 0; t < 4; ++t) {
      const _Float16* xr = x0 + (size_t)t * 16 * Kp;
      bb.q[0] = *(const uint4*)(xr);
      bb.q[1] = *(const uint4*)(xr + 16);
      v8f* accp = (t == 0) ? &acc0 : (t == 1) ? &acc1 : (t == 2) ? &acc2 : &acc3;
      *accp = __builtin_amdgcn_wmma_f32_16x16x32_f16(
          false, a.v, false, bb.v, (short)0, *accp, false, false);
    }
  }
  // store: VGPR r -> (M=r or r+8 depending on lane half, N=lane%16)
  v8f accs[4] = {acc0, acc1, acc2, acc3};
#pragma unroll
  for (int t = 0; t < 4; ++t) {
    int m = m0 + t * 16 + ln;
#pragma unroll
    for (int r = 0; r < 8; ++r) {
      int c = c0 + ((lane < 16) ? r : (r + 8));
      Y[(size_t)c * M + m] = accs[t][r];
    }
  }
}

// ---------------------------------------------------------------------------
// BN statistics (training-mode, population var): block per channel, fixed-tree
// reduction (deterministic). Emits fused scale/shift.
// ---------------------------------------------------------------------------
__global__ void bn_stats_kernel(const float* __restrict__ Y, int M,
                                const float* __restrict__ g,
                                const float* __restrict__ b,
                                float* __restrict__ scale,
                                float* __restrict__ shift) {
  __shared__ float ssum[256], ssq[256];
  int c = blockIdx.x, tid = threadIdx.x;
  float s = 0.f, s2 = 0.f;
  const float* row = Y + (size_t)c * M;
  for (int m = tid; m < M; m += 256) { float v = row[m]; s += v; s2 += v * v; }
  ssum[tid] = s; ssq[tid] = s2;
  __syncthreads();
  for (int o = 128; o > 0; o >>= 1) {
    if (tid < o) { ssum[tid] += ssum[tid + o]; ssq[tid] += ssq[tid + o]; }
    __syncthreads();
  }
  if (tid == 0) {
    float mean = ssum[0] / (float)M;
    float var = ssq[0] / (float)M - mean * mean;
    float sc = g[c] * rsqrtf(var + BN_EPS);
    scale[c] = sc;
    shift[c] = b[c] - mean * sc;
  }
}

// normalize + relu -> f16 X rows (M, C) for the next conv
__global__ void build_x_kernel(const float* __restrict__ Y,
                               const float* __restrict__ scale,
                               const float* __restrict__ shift,
                               int C, int M, _Float16* __restrict__ Xh) {
  size_t idx = (size_t)blockIdx.x * blockDim.x + threadIdx.x;
  if (idx >= (size_t)C * M) return;
  int m = (int)(idx / C), c = (int)(idx % C);
  float v = Y[(size_t)c * M + m] * scale[c] + shift[c];
  Xh[idx] = (_Float16)(v > 0.f ? v : 0.f);
}

// normalize + relu + max over nsample -> pooled feats (B,C,S)
__global__ void pool_kernel(const float* __restrict__ Y,
                            const float* __restrict__ scale,
                            const float* __restrict__ shift,
                            int C, int S, int ns, float* __restrict__ out) {
  int idx = blockIdx.x * blockDim.x + threadIdx.x;
  if (idx >= 8 * C * S) return;
  int b = idx / (C * S), r = idx % (C * S);
  int c = r / S, s = r % S;
  int M = 8 * S * ns;
  float sc = scale[c], sh = shift[c];
  float mv = 0.f; // relu output is >= 0 and ns >= 1
  const float* row = Y + (size_t)c * M + (size_t)(b * S + s) * ns;
  for (int j = 0; j < ns; ++j) {
    float v = row[j] * sc + sh;
    v = v > 0.f ? v : 0.f;
    mv = v > mv ? v : mv;
  }
  out[((size_t)b * C + c) * S + s] = mv;
}

// ---------------------------------------------------------------------------
// FP module input build: 3-NN inverse-distance interpolation + concat skip.
// One thread per (b, n). Writes f16 X row [skip(Cs), interp(Ck), pad].
// ---------------------------------------------------------------------------
__global__ void fp_build_kernel(const float* __restrict__ unknown,
                                const float* __restrict__ known,
                                const float* __restrict__ skipf,
                                const float* __restrict__ knownf,
                                int Nu, int Sk, int Cs, int Ck, int Kp,
                                _Float16* __restrict__ Xh) {
  int q = blockIdx.x * blockDim.x + threadIdx.x;
  if (q >= 8 * Nu) return;
  int b = q / Nu, n = q % Nu;
  const float* u = unknown + (size_t)q * 3;
  float ux = u[0], uy = u[1], uz = u[2];
  const float* kb = known + (size_t)b * Sk * 3;
  float b0 = 3e38f, b1 = 3e38f, b2 = 3e38f;
  int i0 = 0, i1 = 0, i2 = 0;
  for (int s = 0; s < Sk; ++s) {
    float dx = kb[s * 3 + 0] - ux, dy = kb[s * 3 + 1] - uy, dz = kb[s * 3 + 2] - uz;
    float d2 = dx * dx + dy * dy + dz * dz;
    if (d2 < b0)      { b2 = b1; i2 = i1; b1 = b0; i1 = i0; b0 = d2; i0 = s; }
    else if (d2 < b1) { b2 = b1; i2 = i1; b1 = d2; i1 = s; }
    else if (d2 < b2) { b2 = d2; i2 = s; }
  }
  float w0 = 1.f / (b0 + 1e-8f), w1 = 1.f / (b1 + 1e-8f), w2 = 1.f / (b2 + 1e-8f);
  float wn = 1.f / (w0 + w1 + w2);
  w0 *= wn; w1 *= wn; w2 *= wn;
  _Float16* row = Xh + (size_t)q * Kp;
  for (int c = 0; c < Cs; ++c)
    row[c] = (_Float16)skipf[((size_t)b * Cs + c) * Nu + n];
  const float* kf = knownf + (size_t)b * Ck * Sk;
  for (int c = 0; c < Ck; ++c) {
    const float* kc = kf + (size_t)c * Sk;
    row[Cs + c] = (_Float16)(w0 * kc[i0] + w1 * kc[i1] + w2 * kc[i2]);
  }
  for (int k = Cs + Ck; k < Kp; ++k) row[k] = (_Float16)0.f;
}

// normalize + relu -> per-level feats (B,C,Nu) f32 (FP output)
__global__ void fp_final_kernel(const float* __restrict__ Y,
                                const float* __restrict__ scale,
                                const float* __restrict__ shift,
                                int C, int Nu, float* __restrict__ out) {
  int idx = blockIdx.x * blockDim.x + threadIdx.x;
  if (idx >= 8 * C * Nu) return;
  int b = idx / (C * Nu), r = idx % (C * Nu);
  int c = r / Nu, n = r % Nu;
  int M = 8 * Nu;
  float v = Y[(size_t)c * M + (size_t)b * Nu + n] * scale[c] + shift[c];
  out[((size_t)b * C + c) * Nu + n] = v > 0.f ? v : 0.f;
}

// normalize + relu -> whole_feats (B,128,N) f32 into d_out+8
__global__ void fc_apply_kernel(const float* __restrict__ Y,
                                const float* __restrict__ scale,
                                const float* __restrict__ shift,
                                int N, float* __restrict__ out) {
  size_t idx = (size_t)blockIdx.x * blockDim.x + threadIdx.x;
  size_t M = (size_t)8 * N;
  if (idx >= (size_t)128 * M) return;
  int c = (int)(idx / M);
  size_t m = idx % M;
  int b = (int)(m / N), n = (int)(m % N);
  float v = Y[(size_t)c * M + m] * scale[c] + shift[c];
  out[((size_t)b * 128 + c) * N + n] = v > 0.f ? v : 0.f;
}

// classifier head: net = whole_feats[:,:,0]; h = leaky_relu([net,q]@c1W.T+c1b);
// logits = h@c2W.T + c2b. One block, 128 threads, loop over batch.
__global__ void head_kernel(const float* __restrict__ wf,
                            const float* __restrict__ dirs1,
                            const float* __restrict__ dirs2,
                            const float* __restrict__ depth,
                            const float* __restrict__ c1W,
                            const float* __restrict__ c1b,
                            const float* __restrict__ c2W,
                            const float* __restrict__ c2b,
                            float* __restrict__ logits, int N) {
  __shared__ float red[128];
  int o = threadIdx.x;
  const float* wr = c1W + (size_t)o * 135;
  for (int b = 0; b < 8; ++b) {
    float acc = c1b[o];
    for (int c = 0; c < 128; ++c)
      acc += wr[c] * wf[((size_t)b * 128 + c) * N + 0];
    acc += wr[128] * dirs1[b * 3 + 0] + wr[129] * dirs1[b * 3 + 1] +
           wr[130] * dirs1[b * 3 + 2];
    acc += wr[131] * dirs2[b * 3 + 0] + wr[132] * dirs2[b * 3 + 1] +
           wr[133] * dirs2[b * 3 + 2];
    acc += wr[134] * depth[b];
    float h = acc > 0.f ? acc : 0.01f * acc;
    red[o] = h * c2W[o];
    __syncthreads();
    for (int w = 64; w > 0; w >>= 1) {
      if (o < w) red[o] += red[o + w];
      __syncthreads();
    }
    if (o == 0) logits[b] = red[0] + c2b[0];
    __syncthreads();
  }
}

// ---------------------------------------------------------------------------
extern "C" void kernel_launch(void* const* d_in, const int* in_sizes, int n_in,
                              void* d_out, int out_size, void* d_ws, size_t ws_size,
                              hipStream_t stream) {
  (void)in_sizes; (void)n_in; (void)out_size; (void)ws_size;
  const int N = 10000, NS = 32;
  const float* pcs   = (const float*)d_in[0];
  const float* dirs1 = (const float*)d_in[1];
  const float* dirs2 = (const float*)d_in[2];
  const float* depth = (const float*)d_in[3];
  // params pytree flattened with jax sorted-dict-key order:
  // c1W,c1b,c2W,c2b, fc{W,b,g}, fp[0..3][layers]{W,b,g}, sa[0..3][layers]{W,b,g}
  int p = 4;
  auto F = [&](int i) { return (const float*)d_in[i]; };
  const float* c1W = F(p++); const float* c1b = F(p++);
  const float* c2W = F(p++); const float* c2b = F(p++);
  const float* fcW = F(p++); const float* fcb = F(p++); const float* fcg = F(p++);
  struct L { const float *W, *b, *g; };
  L fpL[4][3]; int fp_nl[4] = {3, 2, 2, 2};
  for (int m = 0; m < 4; ++m)
    for (int l = 0; l < fp_nl[m]; ++l) { fpL[m][l].W = F(p++); fpL[m][l].b = F(p++); fpL[m][l].g = F(p++); }
  L saL[4][3];
  for (int m = 0; m < 4; ++m)
    for (int l = 0; l < 3; ++l) { saL[m][l].W = F(p++); saL[m][l].b = F(p++); saL[m][l].g = F(p++); }

  // ---- workspace carve ----
  char* ws = (char*)d_ws;
  size_t off = 0;
  auto carve = [&](size_t bytes) -> void* {
    void* ptr = ws + off;
    off += (bytes + 255) & ~(size_t)255;
    return ptr;
  };
  float* feats0 = (float*)carve((size_t)8 * 3 * N * 4);
  float* xyzw[5]; xyzw[0] = nullptr;
  int Ns[5] = {N, 1024, 256, 64, 16};
  for (int i = 1; i <= 4; ++i) xyzw[i] = (float*)carve((size_t)8 * Ns[i] * 3 * 4);
  float* featsw[5];
  featsw[0] = feats0;
  int fcap[5] = {3, 128, 256, 256, 512};
  for (int i = 1; i <= 4; ++i) featsw[i] = (float*)carve((size_t)8 * fcap[i] * Ns[i] * 4);
  _Float16* WH = (_Float16*)carve((size_t)256 * 768 * 2);       // largest packed W
  float* scale = (float*)carve(512 * 4);
  float* shift = (float*)carve(512 * 4);
  _Float16* XH = (_Float16*)carve((size_t)80000 * 160 * 2);     // largest f16 X
  float* Ybuf = (float*)carve((size_t)64 * 262144 * 4);         // largest conv output

  const float* xyzp[5];
  xyzp[0] = pcs;
  for (int i = 1; i <= 4; ++i) xyzp[i] = xyzw[i];

  auto run_conv = [&](const L& l, int cin, int cout, int Kp, int M) {
    int wn = cout * Kp;
    pack_w_kernel<<<(wn + 255) / 256, 256, 0, stream>>>(l.W, cout, cin, Kp, WH);
    dim3 gg(M / 64, cout / 16);
    wmma_gemm_kernel<<<gg, 32, 0, stream>>>(WH, XH, Ybuf, cout, Kp, M);
    bn_stats_kernel<<<cout, 256, 0, stream>>>(Ybuf, M, l.g, l.b, scale, shift);
  };
  auto relu_to_x = [&](int C, int M) {
    size_t tot = (size_t)C * M;
    build_x_kernel<<<(unsigned)((tot + 255) / 256), 256, 0, stream>>>(Ybuf, scale, shift, C, M, XH);
  };

  // initial channel-first features (== xyz)
  feats0_kernel<<<(8 * N + 255) / 256, 256, 0, stream>>>(pcs, feats0, N);

  // ---- SA modules ----
  float rad2[4] = {0.01f, 0.04f, 0.16f, 0.64f};
  int sa_dims[4][4] = {{6, 32, 32, 64}, {67, 64, 64, 128},
                       {131, 128, 128, 256}, {259, 256, 256, 512}};
  int sa_kp[4] = {32, 96, 160, 288};
  int fch[5] = {3, 64, 128, 256, 512};
  for (int i = 0; i < 4; ++i) {
    int Nprev = Ns[i], S = Ns[i + 1];
    size_t shmem = (size_t)Nprev * 4 + 256 * 4 + 256 * 4;
    fps_kernel<<<8, 256, shmem, stream>>>(xyzp[i], Nprev, S, xyzw[i + 1]);
    int Cf = (i == 0) ? 3 : sa_dims[i - 1][3];
    int Kp = sa_kp[i];
    int nq = 8 * S;
    ball_group_kernel<<<(nq + 63) / 64, 64, 0, stream>>>(
        xyzp[i], xyzp[i + 1], featsw[i], Nprev, S, NS, rad2[i], Cf, Kp, XH);
    int M = 8 * S * NS;
    int d0 = sa_dims[i][1], d1 = sa_dims[i][2], d2 = sa_dims[i][3];
    run_conv(saL[i][0], sa_dims[i][0], d0, Kp, M); relu_to_x(d0, M);
    run_conv(saL[i][1], d0, d1, d0, M);            relu_to_x(d1, M);
    run_conv(saL[i][2], d1, d2, d1, M);
    int pt = 8 * d2 * S;
    pool_kernel<<<(pt + 255) / 256, 256, 0, stream>>>(Ybuf, scale, shift, d2, S, NS, featsw[i + 1]);
  }

  // ---- FP modules (level 4 -> 1) ----
  int fp_dims[4][4] = {{131, 128, 128, 128}, {320, 256, 128, 0},
                       {384, 256, 256, 0}, {768, 256, 256, 0}};
  int fp_kp0[4] = {160, 320, 384, 768};
  for (int i = 4; i >= 1; --i) {
    int m = i - 1;
    int Nu = Ns[i - 1], Sk = Ns[i];
    int Cs = fch[i - 1], Ck = fch[i];
    int Kp = fp_kp0[m];
    int nq = 8 * Nu;
    fp_build_kernel<<<(nq + 127) / 128, 128, 0, stream>>>(
        xyzp[i - 1], xyzp[i], featsw[i - 1], featsw[i], Nu, Sk, Cs, Ck, Kp, XH);
    int M = 8 * Nu;
    int nl = fp_nl[m];
    int kcur = Kp;
    for (int l = 0; l < nl; ++l) {
      int cout = fp_dims[m][l + 1];
      run_conv(fpL[m][l], (l == 0) ? fp_dims[m][0] : fp_dims[m][l], cout, kcur, M);
      if (l + 1 < nl) { relu_to_x(cout, M); kcur = cout; }
    }
    int outc = fp_dims[m][nl];
    if (i > 1) {
      int tt = 8 * outc * Nu;
      fp_final_kernel<<<(tt + 255) / 256, 256, 0, stream>>>(Ybuf, scale, shift, outc, Nu, featsw[i - 1]);
      fch[i - 1] = outc;
    } else {
      relu_to_x(outc, M);   // f16 X for final fc conv
    }
  }

  // ---- fc conv -> whole_feats, then head ----
  {
    L fc{fcW, fcb, fcg};
    int M = 8 * N;
    run_conv(fc, 128, 128, 128, M);
    float* wf = (float*)d_out + 8;
    size_t tot = (size_t)128 * M;
    fc_apply_kernel<<<(unsigned)((tot + 255) / 256), 256, 0, stream>>>(Ybuf, scale, shift, N, wf);
    head_kernel<<<1, 128, 0, stream>>>(wf, dirs1, dirs2, depth, c1W, c1b, c2W, c2b,
                                       (float*)d_out, N);
  }
}
